// CausalSelfAttention_25821343383729
// MI455X (gfx1250) — compile-verified, matrix operand order checked by reading, not run on hardware
//
#include <hip/hip_runtime.h>
#include <hip/hip_bf16.h>

#define BB 4
#define TT 1024
#define CC 1024
#define HH 16
#define HKV 4
#define DD 64
#define GATECH 12
#define NQT (TT / 16)
#define SCOLS 288   // max 272 keys (window 256 + 16), padded to a multiple of 32

typedef __bf16 bf16;
typedef __attribute__((ext_vector_type(8)))  __bf16 v8bf;
typedef __attribute__((ext_vector_type(16))) __bf16 v16bf;
typedef __attribute__((ext_vector_type(8)))  float  v8f;

static __device__ __forceinline__ v16bf combine16(v8bf lo, v8bf hi) {
  v16bf r;
#pragma unroll
  for (int i = 0; i < 8; ++i) { r[i] = lo[i]; r[i + 8] = hi[i]; }
  return r;
}

// A-fragment (16x32, row = lane&15) or B-fragment (32x16, col = lane&15).
// rowptr points at this lane's row/column at contraction offset 0 of the
// current 32-wide K chunk. Lanes 0-15 take K {0..7,16..23}; lanes 16-31 take
// K {8..15,24..31} (two contiguous 16-byte loads).
static __device__ __forceinline__ v16bf load_frag(const bf16* rowptr, int lane) {
  const int base = (lane & 16) ? 8 : 0;
  v8bf lo = *(const v8bf*)(rowptr + base);
  v8bf hi = *(const v8bf*)(rowptr + base + 16);
  return combine16(lo, hi);
}

static __device__ __forceinline__ v8f wmma_bf16(v16bf a, v16bf b, v8f c) {
  return __builtin_amdgcn_wmma_f32_16x16x32_bf16(false, a, false, b, (short)0, c,
                                                 false, false);
}

// async global -> LDS copy of one 16-byte chunk (per lane), ASYNCcnt-tracked
static __device__ __forceinline__ void async_copy_b128(const bf16* gaddr,
                                                       bf16* lds_ptr) {
  unsigned lds_off = (unsigned)(uintptr_t)lds_ptr;           // low 32 bits = LDS byte offset
  unsigned long long ga = (unsigned long long)(uintptr_t)gaddr;
  asm volatile("global_load_async_to_lds_b128 %0, %1, off"
               :: "v"(lds_off), "v"(ga) : "memory");
}

// ---------------------------------------------------------------- conversions
__global__ void cvt_f32_bf16(const float* __restrict__ src, bf16* __restrict__ dst,
                             int n) {
  int i = blockIdx.x * blockDim.x + threadIdx.x;
  int stride = gridDim.x * blockDim.x;
  for (; i < n; i += stride) dst[i] = (bf16)src[i];
}

// ------------------------------------------------------------------ WMMA GEMM
// C[M,N] = A[M,K] (row-major bf16) * W[N,K]^T (row-major bf16).
// 128 threads = 4 waves; block tile 64x64; double-buffered async LDS staging.
template <bool OUTF32>
__global__ void gemm_bf16_wmma(const bf16* __restrict__ A, const bf16* __restrict__ W,
                               void* __restrict__ Cout, int M, int N, int K) {
  __shared__ __align__(16) bf16 As[2][64 * 32];
  __shared__ __align__(16) bf16 Ws[2][64 * 32];

  const int tid  = threadIdx.x;
  const int lane = tid & 31;
  const int wave = tid >> 5;
  const int m0 = blockIdx.y * 64;
  const int n0 = blockIdx.x * 64;
  const int KT = K >> 5;

  // stage one 64x32 A tile + 64x32 W tile: 256 chunks each of 16B; 4 chunks/thread
  auto issue_stage = [&](int buf, int k0) {
#pragma unroll
    for (int it = 0; it < 2; ++it) {
      int chunk = it * 128 + tid;        // 0..255
      int row = chunk >> 2;              // 0..63
      int col = (chunk & 3) * 8;         // element offset in the 32-wide row
      async_copy_b128(A + (size_t)(m0 + row) * K + k0 + col,
                      &As[buf][row * 32 + col]);
      async_copy_b128(W + (size_t)(n0 + row) * K + k0 + col,
                      &Ws[buf][row * 32 + col]);
    }
  };

  v8f acc[4];
#pragma unroll
  for (int i = 0; i < 4; ++i) acc[i] = (v8f)0.0f;

  const int arow = wave * 16 + (lane & 15);
  const int nlr  = lane & 15;

  issue_stage(0, 0);
  for (int kt = 0; kt < KT; ++kt) {
    if (kt + 1 < KT) {
      issue_stage((kt + 1) & 1, (kt + 1) << 5);
      asm volatile("s_wait_asynccnt 0x4" ::: "memory");  // current stage landed
    } else {
      asm volatile("s_wait_asynccnt 0x0" ::: "memory");
    }
    __syncthreads();                                     // all waves' copies visible

    v16bf afrag = load_frag(&As[kt & 1][arow * 32], lane);
#pragma unroll
    for (int nt = 0; nt < 4; ++nt) {
      v16bf bfrag = load_frag(&Ws[kt & 1][(nt * 16 + nlr) * 32], lane);
      acc[nt] = wmma_bf16(afrag, bfrag, acc[nt]);
    }
    __syncthreads();     // everyone done reading buf before it is re-filled
  }

  const int mo = (lane & 16) ? 8 : 0;
  const int mw = m0 + wave * 16;
#pragma unroll
  for (int nt = 0; nt < 4; ++nt) {
#pragma unroll
    for (int r = 0; r < 8; ++r) {
      size_t idx = (size_t)(mw + mo + r) * N + (n0 + nt * 16 + nlr);
      if (OUTF32) ((float*)Cout)[idx] = acc[nt][r];
      else        ((bf16*)Cout)[idx]  = (bf16)acc[nt][r];
    }
  }
}

// ------------------------------------------------------- rotary + RMS epilogue
// One wave per (b, t, head): lane i owns the rotary pair (i, i+32) of D=64.
__global__ void rope_rms_kernel(bf16* __restrict__ qk, const float* __restrict__ cosT,
                                const float* __restrict__ sinT, int nh) {
  const int lane = threadIdx.x & 31;
  const int wave = (blockIdx.x * blockDim.x + threadIdx.x) >> 5;
  const int h = wave % nh;
  const int t = (wave / nh) % TT;
  const int b = wave / (nh * TT);
  if (b >= BB) return;
  const size_t base = (((size_t)b * TT + t) * nh + h) * DD;

  float x1 = (float)qk[base + lane];
  float x2 = (float)qk[base + 32 + lane];
  float c = cosT[t * 32 + lane];
  float s = sinT[t * 32 + lane];
  float y1 = x1 * c + x2 * s;
  float y2 = -x1 * s + x2 * c;

  float ss = y1 * y1 + y2 * y2;
#pragma unroll
  for (int m = 16; m > 0; m >>= 1) ss += __shfl_xor(ss, m, 32);
  float scale = rsqrtf(ss * (1.0f / 64.0f) + 1.1920929e-07f) * 1.2f;

  qk[base + lane]      = (bf16)(y1 * scale);
  qk[base + 32 + lane] = (bf16)(y2 * scale);
}

// ----------------------------------------------------------- gated V epilogue
__global__ void v_gate_kernel(bf16* __restrict__ v, const float* __restrict__ x,
                              const float* __restrict__ ve,
                              const float* __restrict__ Wg) {
  int idx = blockIdx.x * blockDim.x + threadIdx.x;
  const int total = BB * TT * HKV * DD;
  if (idx >= total) return;
  int g = (idx >> 6) & (HKV - 1);
  int bt = idx >> 8;
  float dot = 0.0f;
#pragma unroll
  for (int j = 0; j < GATECH; ++j) dot += x[(size_t)bt * CC + j] * Wg[g * GATECH + j];
  float gate = 3.0f / (1.0f + __expf(-dot));
  v[idx] = (bf16)((float)v[idx] + gate * ve[idx]);
}

// ------------------------------------------------------------------ attention
// One wave per (b, h, 16-query tile). Scores and probs staged in LDS.
__global__ void attn_kernel(const bf16* __restrict__ q, const bf16* __restrict__ k,
                            const bf16* __restrict__ v, bf16* __restrict__ y,
                            const int* __restrict__ win_ptr) {
  __shared__ __align__(16) float S[16 * SCOLS];
  __shared__ __align__(16) bf16  P[16 * SCOLS];
  __shared__ __align__(16) bf16  Vt[DD * 32];   // transposed V chunk: [dim][key]

  const int lane = threadIdx.x & 31;
  int id = blockIdx.x;
  const int qt = id % NQT; id /= NQT;
  const int h  = id % HH;  id /= HH;
  const int b  = id;
  const int g  = h >> 2;                // kv group (rep = 4)
  const int q0 = qt * 16;
  const int win = *win_ptr;

  int k_start = q0 - win;
  if (k_start < 0) k_start = 0;
  k_start &= ~15;
  int nkeys = q0 + 16 - k_start;
  if (nkeys > SCOLS) nkeys = SCOLS;
  const int nkt = nkeys >> 4;

  const int mo = (lane & 16) ? 8 : 0;
  const int nl = lane & 15;

  // Q fragments: row m = lane&15 -> t = q0+m; stride between rows = HH*DD.
  const bf16* qrow = q + (((size_t)b * TT + q0 + nl) * HH + h) * DD;
  v16bf qf0 = load_frag(qrow, lane);        // dims  0..31
  v16bf qf1 = load_frag(qrow + 32, lane);   // dims 32..63

  // ---- scores S = (Q K^T) * scale, masked ----
  for (int kt = 0; kt < nkt; ++kt) {
    const int kb = k_start + kt * 16;
    const bf16* krow = k + (((size_t)b * TT + kb + nl) * HKV + g) * DD;
    v8f acc = (v8f)0.0f;
    acc = wmma_bf16(qf0, load_frag(krow, lane), acc);
    acc = wmma_bf16(qf1, load_frag(krow + 32, lane), acc);
#pragma unroll
    for (int r = 0; r < 8; ++r) {
      int m = mo + r;
      int qpos = q0 + m;
      int kpos = kb + nl;
      float sv = acc[r] * 0.125f;           // 1/sqrt(64)
      if (!(kpos <= qpos && qpos - kpos <= win)) sv = -1e30f;
      S[m * SCOLS + kt * 16 + nl] = sv;
    }
  }
  __syncthreads();

  // ---- softmax per query row (lanes 0..15 each own one row) ----
  if (lane < 16) {
    float* srow = S + lane * SCOLS;
    float mx = -1e30f;
    for (int c = 0; c < nkeys; ++c) mx = fmaxf(mx, srow[c]);
    float sum = 0.0f;
    for (int c = 0; c < nkeys; ++c) sum += __expf(srow[c] - mx);
    float inv = 1.0f / sum;
    bf16* prow = P + lane * SCOLS;
    for (int c = 0; c < nkeys; ++c) prow[c] = (bf16)(__expf(srow[c] - mx) * inv);
    for (int c = nkeys; c < SCOLS; ++c) prow[c] = (bf16)0.0f;
  }
  __syncthreads();

  // ---- O = P V  (contraction over keys in chunks of 32) ----
  v8f oacc[4];
#pragma unroll
  for (int i = 0; i < 4; ++i) oacc[i] = (v8f)0.0f;

  for (int cc = 0; cc < SCOLS / 32; ++cc) {
    // stage transposed V chunk: lane owns key index cc*32+lane
    const int kk = cc * 32 + lane;
    if (kk < nkeys) {
      const bf16* vrow = v + (((size_t)b * TT + k_start + kk) * HKV + g) * DD;
#pragma unroll
      for (int d2 = 0; d2 < DD; ++d2) Vt[d2 * 32 + lane] = vrow[d2];
    } else {
#pragma unroll
      for (int d2 = 0; d2 < DD; ++d2) Vt[d2 * 32 + lane] = (bf16)0.0f;
    }
    __syncthreads();

    v16bf pfrag = load_frag((const bf16*)P + nl * SCOLS + cc * 32, lane);
#pragma unroll
    for (int dg = 0; dg < 4; ++dg) {
      v16bf vfrag = load_frag((const bf16*)Vt + (dg * 16 + nl) * 32, lane);
      oacc[dg] = wmma_bf16(pfrag, vfrag, oacc[dg]);
    }
    __syncthreads();
  }

  bf16* ybase = y + (((size_t)b * TT + q0) * HH + h) * DD;
#pragma unroll
  for (int dg = 0; dg < 4; ++dg)
#pragma unroll
    for (int r = 0; r < 8; ++r)
      ybase[(size_t)(mo + r) * HH * DD + dg * 16 + nl] = (bf16)oacc[dg][r];
}

// -------------------------------------------------------------------- launch
extern "C" void kernel_launch(void* const* d_in, const int* in_sizes, int n_in,
                              void* d_out, int out_size, void* d_ws, size_t ws_size,
                              hipStream_t stream) {
  (void)in_sizes; (void)n_in; (void)out_size; (void)ws_size;
  const float* x    = (const float*)d_in[0];
  const float* ve   = (const float*)d_in[1];
  const float* cosT = (const float*)d_in[2];
  const float* sinT = (const float*)d_in[3];
  const float* Wq   = (const float*)d_in[4];
  const float* Wk   = (const float*)d_in[5];
  const float* Wv   = (const float*)d_in[6];
  const float* Wo   = (const float*)d_in[7];
  const float* Wg   = (const float*)d_in[8];
  const int*   win  = (const int*)d_in[9];
  float* out = (float*)d_out;

  char* ws = (char*)d_ws;
  size_t off = 0;
  auto alloc = [&](size_t bytes) -> char* {
    char* p = ws + off;
    off += (bytes + 255) & ~(size_t)255;
    return p;
  };
  bf16* xb  = (bf16*)alloc((size_t)BB * TT * CC * 2);
  bf16* Wqb = (bf16*)alloc((size_t)HH * DD * CC * 2);
  bf16* Wkb = (bf16*)alloc((size_t)HKV * DD * CC * 2);
  bf16* Wvb = (bf16*)alloc((size_t)HKV * DD * CC * 2);
  bf16* Wob = (bf16*)alloc((size_t)CC * CC * 2);
  bf16* qb  = (bf16*)alloc((size_t)BB * TT * HH * DD * 2);
  bf16* kb  = (bf16*)alloc((size_t)BB * TT * HKV * DD * 2);
  bf16* vb  = (bf16*)alloc((size_t)BB * TT * HKV * DD * 2);
  bf16* yb  = (bf16*)alloc((size_t)BB * TT * HH * DD * 2);

  cvt_f32_bf16<<<(BB * TT * CC + 1023) / 1024, 256, 0, stream>>>(x, xb, BB * TT * CC);
  cvt_f32_bf16<<<(HH * DD * CC + 1023) / 1024, 256, 0, stream>>>(Wq, Wqb, HH * DD * CC);
  cvt_f32_bf16<<<(HKV * DD * CC + 1023) / 1024, 256, 0, stream>>>(Wk, Wkb, HKV * DD * CC);
  cvt_f32_bf16<<<(HKV * DD * CC + 1023) / 1024, 256, 0, stream>>>(Wv, Wvb, HKV * DD * CC);
  cvt_f32_bf16<<<(CC * CC + 1023) / 1024, 256, 0, stream>>>(Wo, Wob, CC * CC);

  dim3 blk(128);
  gemm_bf16_wmma<false><<<dim3(HH * DD / 64, BB * TT / 64), blk, 0, stream>>>(
      xb, Wqb, qb, BB * TT, HH * DD, CC);
  gemm_bf16_wmma<false><<<dim3(HKV * DD / 64, BB * TT / 64), blk, 0, stream>>>(
      xb, Wkb, kb, BB * TT, HKV * DD, CC);
  gemm_bf16_wmma<false><<<dim3(HKV * DD / 64, BB * TT / 64), blk, 0, stream>>>(
      xb, Wvb, vb, BB * TT, HKV * DD, CC);

  v_gate_kernel<<<(BB * TT * HKV * DD + 255) / 256, 256, 0, stream>>>(vb, x, ve, Wg);
  rope_rms_kernel<<<(BB * TT * HH * 32) / 256, 256, 0, stream>>>(qb, cosT, sinT, HH);
  rope_rms_kernel<<<(BB * TT * HKV * 32) / 256, 256, 0, stream>>>(kb, cosT, sinT, HKV);

  attn_kernel<<<BB * HH * NQT, 32, 0, stream>>>(qb, kb, vb, yb, win);

  gemm_bf16_wmma<true><<<dim3(CC / 64, BB * TT / 64), blk, 0, stream>>>(
      yb, Wob, out, BB * TT, CC, CC);
}
